// CogVideoXProcessorAdapter_67688684585346
// MI455X (gfx1250) — compile-verified, matrix operand order checked by reading, not run on hardware
//
#include <hip/hip_runtime.h>
#include <stdint.h>

// ---- problem constants ----
#define B_    2
#define S_TXT 226
#define S_VID 2048
#define S_    2274          // S_TXT + S_VID
#define D_    1024
#define H_    16
#define DH_   64
#define SP_   2304          // S padded so Vt rows stay 16B-aligned
#define M_    (B_ * S_)     // 4548 rows of the concatenated activation matrix

typedef __bf16 v16bf __attribute__((ext_vector_type(16)));
typedef float  v8f   __attribute__((ext_vector_type(8)));

union Frag {
    v16bf v;
    uint4 q[2];
    unsigned short us[16];
};

__device__ __forceinline__ unsigned short f2bf(float f) {
    union { float f; unsigned u; } c; c.f = f;
    unsigned u = c.u;
    unsigned r = (u + 0x7FFFu + ((u >> 16) & 1u)) >> 16;  // RNE
    return (unsigned short)r;
}

__device__ __forceinline__ v8f wmma_bf16(const Frag& a, const Frag& b, v8f c) {
    return __builtin_amdgcn_wmma_f32_16x16x32_bf16(
        /*neg_a=*/false, a.v, /*neg_b=*/false, b.v,
        /*c_mod=*/(short)0, c, /*reuse_a=*/false, /*reuse_b=*/false);
}

__device__ __forceinline__ void load_frag(Frag& f, const unsigned short* p, int kb) {
    f.q[0] = *(const uint4*)(p + kb);
    f.q[1] = *(const uint4*)(p + kb + 16);
}

// async 16-byte copy global -> LDS (ASYNCcnt-tracked, CDNA5 path)
__device__ __forceinline__ void async_cp16(const unsigned short* gptr, const unsigned short* lptr) {
    unsigned long long ga = (unsigned long long)(uintptr_t)gptr;
    unsigned lo = (unsigned)(uintptr_t)lptr;   // low 32 bits of flat addr == LDS byte offset
    asm volatile("global_load_async_to_lds_b128 %0, %1, off" :: "v"(lo), "v"(ga) : "memory");
}
__device__ __forceinline__ void wait_async0() {
    asm volatile("s_wait_asynccnt 0x0" ::: "memory");
}

// ---------------------------------------------------------------------------
// Kernel 0: concat text+video activations and convert everything to bf16
// ---------------------------------------------------------------------------
__global__ void convert_kernel(const float* __restrict__ hid,
                               const float* __restrict__ enc,
                               const float* __restrict__ wq, const float* __restrict__ wk,
                               const float* __restrict__ wv, const float* __restrict__ wo,
                               unsigned short* __restrict__ xbf,
                               unsigned short* __restrict__ wqb, unsigned short* __restrict__ wkb,
                               unsigned short* __restrict__ wvb, unsigned short* __restrict__ wob) {
    long i = (long)blockIdx.x * blockDim.x + threadIdx.x;
    const long NX = (long)M_ * D_;
    const long NW = (long)D_ * D_;
    if (i < NX) {
        long m = i / D_, d = i % D_;
        int b = (int)(m / S_), s = (int)(m % S_);
        float v = (s < S_TXT) ? enc[((long)b * S_TXT + s) * D_ + d]
                              : hid[((long)b * S_VID + (s - S_TXT)) * D_ + d];
        xbf[i] = f2bf(v);
    } else if (i < NX + 4 * NW) {
        long j = i - NX;
        int  w = (int)(j / NW);
        long k = j % NW;
        const float* src = (w == 0) ? wq : (w == 1) ? wk : (w == 2) ? wv : wo;
        unsigned short* dst = (w == 0) ? wqb : (w == 1) ? wkb : (w == 2) ? wvb : wob;
        dst[k] = f2bf(src[k]);
    }
}

// ---------------------------------------------------------------------------
// GEMM: Y[m,n] = sum_d X[m,d] * W[n,d] + bias[n]
// One wave computes a 16(M) x 64(N) tile; K loop in 32-wide bf16 WMMA steps.
// mode 0: store fp32 to rawOut[m*D+n]   (QKV projections)
// mode 1: store fp32 into the split (video, text) output layout (out proj)
// ---------------------------------------------------------------------------
__global__ void gemm_bf16_kernel(const unsigned short* __restrict__ X,
                                 const unsigned short* __restrict__ W,
                                 const float* __restrict__ bias,
                                 float* __restrict__ rawOut,
                                 float* __restrict__ splitOut,
                                 int mode) {
    const int lane = threadIdx.x & 31;
    const int l15  = lane & 15;
    const int half = lane >> 4;
    const int kb   = half ? 8 : 0;
    const int m0   = blockIdx.x * 16;
    const int n0   = blockIdx.y * 64;

    int mrow = m0 + l15; if (mrow >= M_) mrow = M_ - 1;
    const unsigned short* xrow = X + (long)mrow * D_;

    v8f acc[4];
#pragma unroll
    for (int t = 0; t < 4; ++t) acc[t] = 0.0f;

    for (int kk = 0; kk < D_; kk += 32) {
        Frag a;
        load_frag(a, xrow + kk, kb);
#pragma unroll
        for (int t = 0; t < 4; ++t) {
            Frag bf;
            load_frag(bf, W + (long)(n0 + t * 16 + l15) * D_ + kk, kb);
            acc[t] = wmma_bf16(a, bf, acc[t]);
        }
    }

#pragma unroll
    for (int t = 0; t < 4; ++t) {
        const int col = n0 + t * 16 + l15;
        const float bv = bias[col];
#pragma unroll
        for (int g = 0; g < 8; ++g) {
            const int row = m0 + g + half * 8;
            if (row < M_) {
                const float val = acc[t][g] + bv;
                if (mode == 0) {
                    rawOut[(long)row * D_ + col] = val;
                } else {
                    const int b = row / S_, s = row % S_;
                    long off = (s >= S_TXT)
                        ? ((long)b * S_VID + (s - S_TXT)) * D_ + col
                        : (long)B_ * S_VID * D_ + ((long)b * S_TXT + s) * D_ + col;
                    splitOut[off] = val;
                }
            }
        }
    }
}

// ---------------------------------------------------------------------------
// Kernel 2: per-(b,s,h) row: LayerNorm over DH=64, RoPE on video tokens,
// pack Q/K bf16 as [B,H,S,DH] and V transposed as [B,H,DH,SP_].
// One wave per row, 2 elements (one rotary pair) per lane.
// ---------------------------------------------------------------------------
__global__ void ln_rope_kernel(const float* __restrict__ Qr, const float* __restrict__ Kr,
                               const float* __restrict__ Vr,
                               const float* __restrict__ cosb, const float* __restrict__ sinb,
                               const float* __restrict__ lnqw, const float* __restrict__ lnqb,
                               const float* __restrict__ lnkw, const float* __restrict__ lnkb,
                               unsigned short* __restrict__ Qb, unsigned short* __restrict__ Kb,
                               unsigned short* __restrict__ Vt) {
    const int lane = threadIdx.x & 31;
    const int wid  = threadIdx.x >> 5;
    const long R = (long)blockIdx.x * (blockDim.x >> 5) + wid;
    if (R >= (long)B_ * S_ * H_) return;
    const int h = (int)(R % H_);
    const int s = (int)((R / H_) % S_);
    const int b = (int)(R / ((long)H_ * S_));

    const int d0 = lane * 2, d1 = d0 + 1;
    const long rawoff = ((long)(b * S_ + s)) * D_ + h * DH_;
    const long dstoff = (((long)(b * H_ + h)) * S_ + s) * DH_;
    const bool rope = (s >= S_TXT);
    float c = 1.0f, sn = 0.0f;
    if (rope) {
        const int vs = s - S_TXT;
        c  = cosb[(long)vs * DH_ + d0];
        sn = sinb[(long)vs * DH_ + d0];
    }

#pragma unroll
    for (int which = 0; which < 2; ++which) {
        const float* src = which ? Kr : Qr;
        const float* lw  = which ? lnkw : lnqw;
        const float* lb  = which ? lnkb : lnqb;
        unsigned short* dst = which ? Kb : Qb;

        float x0 = src[rawoff + d0], x1 = src[rawoff + d1];
        float sum = x0 + x1, sq = x0 * x0 + x1 * x1;
#pragma unroll
        for (int mm = 16; mm >= 1; mm >>= 1) {
            sum += __shfl_xor(sum, mm, 32);
            sq  += __shfl_xor(sq,  mm, 32);
        }
        const float mu  = sum * (1.0f / 64.0f);
        const float var = sq * (1.0f / 64.0f) - mu * mu;
        const float rs  = rsqrtf(var + 1e-5f);
        const float y0 = (x0 - mu) * rs * lw[d0] + lb[d0];
        const float y1 = (x1 - mu) * rs * lw[d1] + lb[d1];
        const float o0 = rope ? (y0 * c - y1 * sn) : y0;
        const float o1 = rope ? (y1 * c + y0 * sn) : y1;
        dst[dstoff + d0] = f2bf(o0);
        dst[dstoff + d1] = f2bf(o1);
    }
    {
        const float v0 = Vr[rawoff + d0], v1 = Vr[rawoff + d1];
        const long vt = (((long)(b * H_ + h)) * DH_) * SP_ + s;
        Vt[vt + (long)d0 * SP_] = f2bf(v0);
        Vt[vt + (long)d1 * SP_] = f2bf(v1);
    }
}

// ---------------------------------------------------------------------------
// Kernel 3: flash attention, 128-thread blocks (4 waves, one 16-row q tile
// per wave). K/V chunks of 32 keys are DMA'd into LDS with
// GLOBAL_LOAD_ASYNC_TO_LDS_B128, double-buffered so the async copy of chunk
// i+1 overlaps the 8 WMMAs + online softmax of chunk i. One barrier per
// chunk after s_wait_asynccnt 0 closes the handoff.
// ---------------------------------------------------------------------------
__global__ void attn_kernel(const unsigned short* __restrict__ Qm,
                            const unsigned short* __restrict__ Km,
                            const unsigned short* __restrict__ Vt,
                            unsigned short* __restrict__ Obf) {
    __shared__ unsigned short sK[2][32 * DH_];   // [buf][key_row][d]    4KB each
    __shared__ unsigned short sV[2][DH_ * 32];   // [buf][d][key_col]    4KB each
    __shared__ unsigned short sP[4][16 * 32];    // per-wave P transpose scratch

    const int tid  = threadIdx.x;
    const int lane = tid & 31;
    const int wid  = tid >> 5;
    const int l15  = lane & 15;
    const int half = (lane >> 4) & 1;
    const int kb   = half ? 8 : 0;

    const int q0 = blockIdx.x * 64 + wid * 16;
    const int h  = blockIdx.y;
    const int b  = blockIdx.z;

    const long bhq = ((long)(b * H_ + h)) * S_;   // row base for Qm/Km
    const long bhv = ((long)(b * H_ + h)) * DH_;  // DH-row base for Vt

    // ---- resident A fragments of the Q tile (two 32-wide halves of DH) ----
    int qrow = q0 + l15; if (qrow >= S_) qrow = S_ - 1;
    const unsigned short* qp = Qm + (bhq + qrow) * DH_;
    Frag aq0, aq1;
    load_frag(aq0, qp, kb);
    load_frag(aq1, qp + 32, kb);

    v8f oacc[4];
#pragma unroll
    for (int t = 0; t < 4; ++t) oacc[t] = 0.0f;
    float mrun[8], lrun[8];
#pragma unroll
    for (int g = 0; g < 8; ++g) { mrun[g] = -1e30f; lrun[g] = 0.0f; }

    const float scale = 0.125f;  // 1/sqrt(DH)
    const int NCH = (S_ + 31) / 32;

    // stage one 32-key K/V chunk into LDS buffer bufi (whole block, async)
    auto stage = [&](int k0, int bufi) {
        // K chunk: rows are contiguous in [S,DH] -> one 4KB linear range
        const unsigned short* kbase = Km + (bhq + k0) * DH_;
        async_cp16(kbase + tid * 8,         &sK[bufi][tid * 8]);
        async_cp16(kbase + (tid + 128) * 8, &sK[bufi][(tid + 128) * 8]);
        // V chunk: 64 DH-rows x 64B from the transposed V
#pragma unroll
        for (int r = 0; r < 2; ++r) {
            const int seg = tid + r * 128;          // 0..255
            const int row = seg >> 2, part = seg & 3;
            async_cp16(Vt + (bhv + row) * SP_ + k0 + part * 8,
                       &sV[bufi][row * 32 + part * 8]);
        }
    };

    stage(0, 0);
    wait_async0();
    __syncthreads();

    for (int ci = 0; ci < NCH; ++ci) {
        const int k0  = ci * 32;
        const int cur = ci & 1;
        if (ci + 1 < NCH) stage(k0 + 32, cur ^ 1);   // overlap with compute

        // ---- scores: S(16x32) = Q(16x64) x K^T, two 16-column groups ----
        v8f c0 = 0.0f, c1 = 0.0f;
        {
            Frag b0, b1;
            load_frag(b0, &sK[cur][l15 * DH_], kb);
            load_frag(b1, &sK[cur][l15 * DH_ + 32], kb);
            c0 = wmma_bf16(aq0, b0, c0);
            c0 = wmma_bf16(aq1, b1, c0);
            load_frag(b0, &sK[cur][(16 + l15) * DH_], kb);
            load_frag(b1, &sK[cur][(16 + l15) * DH_ + 32], kb);
            c1 = wmma_bf16(aq0, b0, c1);
            c1 = wmma_bf16(aq1, b1, c1);
        }

        const bool ok0 = (k0 + l15) < S_;
        const bool ok1 = (k0 + 16 + l15) < S_;

        // ---- online softmax (rows live across the 16-lane halves) ----
#pragma unroll
        for (int g = 0; g < 8; ++g) {
            float s0 = ok0 ? c0[g] * scale : -1e30f;
            float s1 = ok1 ? c1[g] * scale : -1e30f;
            float rmax = fmaxf(s0, s1);
#pragma unroll
            for (int mm = 1; mm < 16; mm <<= 1)
                rmax = fmaxf(rmax, __shfl_xor(rmax, mm, 32));
            const float mnew = fmaxf(mrun[g], rmax);
            const float cf = __expf(mrun[g] - mnew);
            const float p0 = __expf(s0 - mnew);
            const float p1 = __expf(s1 - mnew);
            float rsum = p0 + p1;
#pragma unroll
            for (int mm = 1; mm < 16; mm <<= 1)
                rsum += __shfl_xor(rsum, mm, 32);
            lrun[g] = lrun[g] * cf + rsum;
            mrun[g] = mnew;
#pragma unroll
            for (int t = 0; t < 4; ++t) oacc[t][g] *= cf;
            const int prow = g + half * 8;
            sP[wid][prow * 32 + l15]      = f2bf(p0);
            sP[wid][prow * 32 + 16 + l15] = f2bf(p1);
        }
        // per-wave scratch: DS ops from the same wave are in order, no barrier

        // ---- reload P in A-matrix layout, then O += P x V ----
        Frag ap;
        ap.q[0] = *(const uint4*)(&sP[wid][l15 * 32 + kb]);
        ap.q[1] = *(const uint4*)(&sP[wid][l15 * 32 + kb + 16]);

#pragma unroll
        for (int t = 0; t < 4; ++t) {
            Frag bv;
            load_frag(bv, &sV[cur][(t * 16 + l15) * 32], kb);
            oacc[t] = wmma_bf16(ap, bv, oacc[t]);
        }

        wait_async0();      // next chunk's DMA landed
        __syncthreads();    // everyone done reading `cur` before it is re-staged
    }

    // ---- normalize and store O as bf16 in [B, S, H, DH] (= [M, D]) ----
#pragma unroll
    for (int t = 0; t < 4; ++t) {
#pragma unroll
        for (int g = 0; g < 8; ++g) {
            const int row = q0 + g + half * 8;
            if (row < S_) {
                const float val = oacc[t][g] / lrun[g];
                Obf[((long)(b * S_ + row)) * D_ + h * DH_ + t * 16 + l15] = f2bf(val);
            }
        }
    }
}

// ---------------------------------------------------------------------------
// host-side launch
// ---------------------------------------------------------------------------
extern "C" void kernel_launch(void* const* d_in, const int* in_sizes, int n_in,
                              void* d_out, int out_size, void* d_ws, size_t ws_size,
                              hipStream_t stream) {
    const float* hid  = (const float*)d_in[0];
    const float* enc  = (const float*)d_in[1];
    const float* cosb = (const float*)d_in[2];
    const float* sinb = (const float*)d_in[3];
    const float* wq   = (const float*)d_in[4];
    const float* bq   = (const float*)d_in[5];
    const float* wk   = (const float*)d_in[6];
    const float* bk   = (const float*)d_in[7];
    const float* wv   = (const float*)d_in[8];
    const float* bv   = (const float*)d_in[9];
    const float* wo   = (const float*)d_in[10];
    const float* bo   = (const float*)d_in[11];
    const float* lnqw = (const float*)d_in[12];
    const float* lnqb = (const float*)d_in[13];
    const float* lnkw = (const float*)d_in[14];
    const float* lnkb = (const float*)d_in[15];

    char* ws = (char*)d_ws;
    size_t off = 0;
    auto alloc = [&](size_t bytes) -> void* {
        off = (off + 255) & ~(size_t)255;
        void* p = ws + off;
        off += bytes;
        return p;
    };

    unsigned short* xbf = (unsigned short*)alloc((size_t)M_ * D_ * 2);
    unsigned short* wqb = (unsigned short*)alloc((size_t)D_ * D_ * 2);
    unsigned short* wkb = (unsigned short*)alloc((size_t)D_ * D_ * 2);
    unsigned short* wvb = (unsigned short*)alloc((size_t)D_ * D_ * 2);
    unsigned short* wob = (unsigned short*)alloc((size_t)D_ * D_ * 2);
    float* qraw = (float*)alloc((size_t)M_ * D_ * 4);
    float* kraw = (float*)alloc((size_t)M_ * D_ * 4);
    float* vraw = (float*)alloc((size_t)M_ * D_ * 4);
    unsigned short* qbb = (unsigned short*)alloc((size_t)M_ * D_ * 2);
    unsigned short* kbb = (unsigned short*)alloc((size_t)M_ * D_ * 2);
    unsigned short* vtb = (unsigned short*)alloc((size_t)B_ * H_ * DH_ * SP_ * 2);
    // Obf aliases qraw (dead after ln_rope_kernel): bf16 needs half its bytes.
    unsigned short* obf = (unsigned short*)qraw;

    // 0) convert/concat
    {
        const long ntot = (long)M_ * D_ + 4L * D_ * D_;
        const int blocks = (int)((ntot + 255) / 256);
        convert_kernel<<<blocks, 256, 0, stream>>>(hid, enc, wq, wk, wv, wo,
                                                   xbf, wqb, wkb, wvb, wob);
    }

    // 1) QKV projections (bf16 WMMA, fp32 out)
    {
        dim3 grid((M_ + 15) / 16, D_ / 64);
        gemm_bf16_kernel<<<grid, 32, 0, stream>>>(xbf, wqb, bq, qraw, nullptr, 0);
        gemm_bf16_kernel<<<grid, 32, 0, stream>>>(xbf, wkb, bk, kraw, nullptr, 0);
        gemm_bf16_kernel<<<grid, 32, 0, stream>>>(xbf, wvb, bv, vraw, nullptr, 0);
    }

    // 2) per-head LN + RoPE + repack
    {
        const long rows = (long)B_ * S_ * H_;
        const int wavesPerBlock = 8;  // blockDim 256 = 8 waves
        const int blocks = (int)((rows + wavesPerBlock - 1) / wavesPerBlock);
        ln_rope_kernel<<<blocks, 32 * wavesPerBlock, 0, stream>>>(
            qraw, kraw, vraw, cosb, sinb, lnqw, lnqb, lnkw, lnkb, qbb, kbb, vtb);
    }

    // 3) flash attention (4 q-tiles per block share async-staged K/V chunks)
    {
        dim3 grid((S_ + 63) / 64, H_, B_);
        attn_kernel<<<grid, 128, 0, stream>>>(qbb, kbb, vtb, obf);
    }

    // 4) output projection, split into (video, text)
    {
        dim3 grid((M_ + 15) / 16, D_ / 64);
        gemm_bf16_kernel<<<grid, 32, 0, stream>>>(obf, wob, bo, nullptr, (float*)d_out, 1);
    }
}
